// AttentionBlock_27788438405807
// MI455X (gfx1250) — matrix operand order repaired and checked by reading, not compiled, not run on hardware
//
#include <hip/hip_runtime.h>
#include <hip/hip_bf16.h>
#include <stdint.h>

#define N_REAL  8192
#define NTOK    8193          // N+1 rows (with CLS row 0)
#define DIM     1024
#define NP      8224          // padded token count: 32 * 257
#define SCALE_Q 0.03125f

typedef __attribute__((ext_vector_type(16))) __bf16  v16bf;
typedef __attribute__((ext_vector_type(8)))  float   v8f;
typedef __attribute__((ext_vector_type(8)))  unsigned int v8u;

static __device__ __forceinline__ v16bf as_bf(v8u u) {
    union { v8u u; v16bf b; } c; c.u = u; return c.b;
}
static __device__ __forceinline__ v8f v8f_zero() {
    v8f z;
#pragma unroll
    for (int i = 0; i < 8; ++i) z[i] = 0.f;
    return z;
}
static __device__ __forceinline__ v8u ld_b256_u16(const uint16_t* p) {
    // two b128 loads -> 8 dwords (16 bf16)
    const uint4 lo = *(const uint4*)(p);
    const uint4 hi = *(const uint4*)(p + 16);
    v8u r; r[0]=lo.x; r[1]=lo.y; r[2]=lo.z; r[3]=lo.w;
           r[4]=hi.x; r[5]=hi.y; r[6]=hi.z; r[7]=hi.w;
    return r;
}
static __device__ __forceinline__ v8u ld_a_frag(const uint16_t* p) {
    // A-fragment: b128 at +0 (K pairs 0..7 of lane half) and b128 at +16 elems
    const uint4 lo = *(const uint4*)(p);
    const uint4 hi = *(const uint4*)(p + 16);
    v8u r; r[0]=lo.x; r[1]=lo.y; r[2]=lo.z; r[3]=lo.w;
           r[4]=hi.x; r[5]=hi.y; r[6]=hi.z; r[7]=hi.w;
    return r;
}
#define WMMA_BF16(a, b, c) \
    __builtin_amdgcn_wmma_f32_16x16x32_bf16(false, (a), false, (b), (short)0, (c), false, false)

// ---------------------------------------------------------------------------
// Kernel 0a: convert x (f32 [NTOK][DIM]) -> xb (bf16 [NP][DIM], pad rows
// replicate row 8192).  8 elements per thread, uint4 stores.
// ---------------------------------------------------------------------------
__global__ __launch_bounds__(256)
void cvt_x_kernel(const float* __restrict__ x, __bf16* __restrict__ xb) {
    const size_t e8 = ((size_t)blockIdx.x * 256 + threadIdx.x) * 8;
    int row = (int)(e8 >> 10);
    int col = (int)(e8 & 1023);
    int sr  = row > N_REAL ? N_REAL : row;
    const float* s = x + (size_t)sr * DIM + col;
    float4 f0 = *(const float4*)(s);
    float4 f1 = *(const float4*)(s + 4);
    union { __bf16 h[8]; uint4 u; } r;
    r.h[0]=(__bf16)f0.x; r.h[1]=(__bf16)f0.y; r.h[2]=(__bf16)f0.z; r.h[3]=(__bf16)f0.w;
    r.h[4]=(__bf16)f1.x; r.h[5]=(__bf16)f1.y; r.h[6]=(__bf16)f1.z; r.h[7]=(__bf16)f1.w;
    *(uint4*)(xb + e8) = r.u;
}

// ---------------------------------------------------------------------------
// Kernel 0b: convert WQ/WK/WV (f32 [DIM][DIM] each) -> Wb (bf16 [3][DIM][DIM])
// ---------------------------------------------------------------------------
__global__ __launch_bounds__(256)
void cvt_w_kernel(const float* __restrict__ WQ, const float* __restrict__ WK,
                  const float* __restrict__ WV, __bf16* __restrict__ Wb) {
    const size_t e8 = ((size_t)blockIdx.x * 256 + threadIdx.x) * 8;
    const size_t msz = (size_t)DIM * DIM;
    int sel = (int)(e8 / msz);
    size_t off = e8 - (size_t)sel * msz;
    const float* W = (sel == 0) ? WQ : ((sel == 1) ? WK : WV);
    float4 f0 = *(const float4*)(W + off);
    float4 f1 = *(const float4*)(W + off + 4);
    union { __bf16 h[8]; uint4 u; } r;
    r.h[0]=(__bf16)f0.x; r.h[1]=(__bf16)f0.y; r.h[2]=(__bf16)f0.z; r.h[3]=(__bf16)f0.w;
    r.h[4]=(__bf16)f1.x; r.h[5]=(__bf16)f1.y; r.h[6]=(__bf16)f1.z; r.h[7]=(__bf16)f1.w;
    *(uint4*)(Wb + e8) = r.u;
}

// ---------------------------------------------------------------------------
// Kernel 1: pure-bf16 QKV GEMM, software-pipelined.
//   Q =  xb @ WQ^T (bf16, [NP][DIM]),  K = xb @ WK^T ([NP][DIM]),
//   Vt = (xb @ WV^T)^T ([DIM][NP]).
// Block 128 = 4 waves; each wave: one 16(M) x 64(N) strip = 4 independent
// C tiles; next chunk's fragments loaded before current chunk's wmmas.
// ---------------------------------------------------------------------------
__global__ __launch_bounds__(128)
void qkv_kernel(const __bf16* __restrict__ xb, const __bf16* __restrict__ Wb,
                __bf16* __restrict__ Qb, __bf16* __restrict__ Kb,
                __bf16* __restrict__ Vt) {
    const int wave = threadIdx.x >> 5;
    const int lane = threadIdx.x & 31;
    const int hl   = lane >> 4;
    const int l16  = lane & 15;
    const int m0   = blockIdx.y * 16;
    const int n0   = blockIdx.x * 256 + wave * 64;
    const int which = blockIdx.z;      // 0=Q, 1=K, 2=V

    const uint16_t* xrow = (const uint16_t*)xb + (size_t)(m0 + l16) * DIM + hl * 8;
    const uint16_t* wbase = (const uint16_t*)Wb + (size_t)which * DIM * DIM
                          + (size_t)(n0 + l16) * DIM + hl * 8;

    v8f c0 = v8f_zero(), c1 = v8f_zero(), c2 = v8f_zero(), c3 = v8f_zero();

    // prologue: fragments for chunk 0
    v8u a_c  = ld_a_frag(xrow);
    v8u b_c0 = ld_b256_u16(wbase);
    v8u b_c1 = ld_b256_u16(wbase + 16 * DIM);
    v8u b_c2 = ld_b256_u16(wbase + 32 * DIM);
    v8u b_c3 = ld_b256_u16(wbase + 48 * DIM);

#pragma unroll 2
    for (int kk = 0; kk < DIM - 32; kk += 32) {
        const int kn = kk + 32;
        // issue next chunk's loads before consuming current fragments
        v8u a_n  = ld_a_frag(xrow + kn);
        v8u b_n0 = ld_b256_u16(wbase + kn);
        v8u b_n1 = ld_b256_u16(wbase + 16 * DIM + kn);
        v8u b_n2 = ld_b256_u16(wbase + 32 * DIM + kn);
        v8u b_n3 = ld_b256_u16(wbase + 48 * DIM + kn);

        const v16bf a = as_bf(a_c);
        c0 = WMMA_BF16(a, as_bf(b_c0), c0);
        c1 = WMMA_BF16(a, as_bf(b_c1), c1);
        c2 = WMMA_BF16(a, as_bf(b_c2), c2);
        c3 = WMMA_BF16(a, as_bf(b_c3), c3);

        a_c = a_n; b_c0 = b_n0; b_c1 = b_n1; b_c2 = b_n2; b_c3 = b_n3;
    }
    {   // epilogue chunk
        const v16bf a = as_bf(a_c);
        c0 = WMMA_BF16(a, as_bf(b_c0), c0);
        c1 = WMMA_BF16(a, as_bf(b_c1), c1);
        c2 = WMMA_BF16(a, as_bf(b_c2), c2);
        c3 = WMMA_BF16(a, as_bf(b_c3), c3);
    }

    v8f ct[4] = {c0, c1, c2, c3};
    if (which == 2) {
#pragma unroll
        for (int t = 0; t < 4; ++t)
#pragma unroll
            for (int v = 0; v < 8; ++v) {
                int m = m0 + v + hl * 8;
                int n = n0 + t * 16 + l16;
                Vt[(size_t)n * NP + m] = (__bf16)ct[t][v];
            }
    } else {
        __bf16* O = (which == 0) ? Qb : Kb;
#pragma unroll
        for (int t = 0; t < 4; ++t)
#pragma unroll
            for (int v = 0; v < 8; ++v) {
                int m = m0 + v + hl * 8;
                O[(size_t)m * DIM + n0 + t * 16 + l16] = (__bf16)ct[t][v];
            }
    }
}

// ---------------------------------------------------------------------------
// Kernel 2: flash attention.
// Block = 512 threads = 16 waves: 4 query sub-tiles (16 rows) x 4 d-slices
// (256 cols) => 64 query rows per block (bounds K/V L2 re-streaming).
// ---------------------------------------------------------------------------
__global__ __launch_bounds__(512)
void attn_kernel(const __bf16* __restrict__ Qb, const __bf16* __restrict__ Kb,
                 const __bf16* __restrict__ Vt,
                 const float* __restrict__ phi3, const float* __restrict__ phis,
                 const float* __restrict__ phie,
                 float* __restrict__ out) {
    __shared__ float   s_bias[64 * 32];          //  8 KB
    __shared__ float   s_part[16 * 512];         // 32 KB  [qt][ds][2*256]
    __shared__ __bf16  s_pt[64 * 32];            //  4 KB  shared P tiles
    __shared__ float   s_m[64], s_l[64], s_alpha[64];

    const int tid  = threadIdx.x;
    const int wave = tid >> 5;
    const int lane = tid & 31;
    const int hl   = lane >> 4;
    const int l16  = tid & 15;
    const int qt   = wave >> 2;                  // query sub-tile 0..3
    const int ds   = wave & 3;                   // d-slice 0..3
    const int q0   = blockIdx.x * 64;
    const int d0   = ds * 256;

    if (tid < 64) { s_m[tid] = -1e30f; s_l[tid] = 0.f; }

    // ---- preload Q fragments: rows q0+qt*16..+15, k-slice [d0, d0+256) ----
    v8u qa[8];
    {
        int qr = q0 + qt * 16 + l16; if (qr > NP - 1) qr = NP - 1;
        const uint16_t* qrow = (const uint16_t*)Qb + (size_t)qr * DIM + d0 + hl * 8;
#pragma unroll
        for (int cc = 0; cc < 8; ++cc) qa[cc] = ld_a_frag(qrow + cc * 32);
    }

    v8f o[16];
#pragma unroll
    for (int t = 0; t < 16; ++t) o[t] = v8f_zero();

    for (int j0 = 0; j0 < NP; j0 += 32) {
        __syncthreads();   // protect LDS reuse from prior iter (also init vis)

        // ---- cooperative bias tile load (64x32), sum of 3 phi matrices ----
        {
            int e = tid * 4;
#pragma unroll
            for (int u = 0; u < 4; ++u, ++e) {
                int m = e >> 5, cc = e & 31;
                int i = q0 + m, j = j0 + cc;
                float b = 0.f;
                if (i >= 1 && i < NTOK && j >= 1 && j < NTOK) {
                    size_t ip = (size_t)(i - 1) * N_REAL + (j - 1);
                    b = phi3[ip] + phis[ip] + phie[ip];
                }
                s_bias[e] = b;
            }
            // prefetch next key-tile's phi rows (one 128B row per leader)
            if ((tid & 7) == 0) {
                int m = tid >> 3;
                int i = q0 + m, jn = j0 + 32;
                if (i >= 1 && i < NTOK && jn < NTOK) {
                    size_t ip = (size_t)(i - 1) * N_REAL + (jn - 1);
                    __builtin_prefetch(&phi3[ip], 0, 0);
                    __builtin_prefetch(&phis[ip], 0, 0);
                    __builtin_prefetch(&phie[ip], 0, 0);
                }
            }
        }

        // ---- partial S (pipelined): 16 rows x 32 keys over this k-slice ----
        v8f s0 = v8f_zero(), s1 = v8f_zero();
        {
            const uint16_t* k0p = (const uint16_t*)Kb + (size_t)(j0 + l16)      * DIM + d0 + hl * 8;
            const uint16_t* k1p = (const uint16_t*)Kb + (size_t)(j0 + 16 + l16) * DIM + d0 + hl * 8;
            v8u b0c = ld_b256_u16(k0p);
            v8u b1c = ld_b256_u16(k1p);
#pragma unroll
            for (int cc = 0; cc < 7; ++cc) {
                v8u b0n = ld_b256_u16(k0p + (cc + 1) * 32);
                v8u b1n = ld_b256_u16(k1p + (cc + 1) * 32);
                const v16bf a = as_bf(qa[cc]);
                s0 = WMMA_BF16(a, as_bf(b0c), s0);
                s1 = WMMA_BF16(a, as_bf(b1c), s1);
                b0c = b0n; b1c = b1n;
            }
            const v16bf a = as_bf(qa[7]);
            s0 = WMMA_BF16(a, as_bf(b0c), s0);
            s1 = WMMA_BF16(a, as_bf(b1c), s1);
        }
        {
            float* bp = s_part + (size_t)(qt * 4 + ds) * 512;
#pragma unroll
            for (int v = 0; v < 8; ++v) {
                bp[v * 32 + lane]       = s0[v];
                bp[256 + v * 32 + lane] = s1[v];
            }
        }
        __syncthreads();

        // ---- distributed softmax: half-wave (tid>>4) owns rows 2h, 2h+1 ----
        {
            const int hw = tid >> 4;
#pragma unroll
            for (int u = 0; u < 2; ++u) {
                const int m   = hw * 2 + u;          // block-local row 0..63
                const int qtm = m >> 4;
                const int r   = m & 15;
                const int v   = r & 7, rh = r >> 3;
                float a0 = 0.f, a1 = 0.f;
#pragma unroll
                for (int dd = 0; dd < 4; ++dd) {
                    const float* bp = s_part + (size_t)(qtm * 4 + dd) * 512
                                    + v * 32 + rh * 16 + l16;
                    a0 += bp[0];
                    a1 += bp[256];
                }
                a0 = fmaf(a0, SCALE_Q, s_bias[m * 32 + l16]);
                a1 = fmaf(a1, SCALE_Q, s_bias[m * 32 + 16 + l16]);
                if (j0 + l16      >= NTOK) a0 = -1e30f;
                if (j0 + 16 + l16 >= NTOK) a1 = -1e30f;

                float rmax = fmaxf(a0, a1);
#pragma unroll
                for (int msk = 1; msk < 16; msk <<= 1)
                    rmax = fmaxf(rmax, __shfl_xor(rmax, msk, 32));
                float mold = s_m[m];
                float mnew = fmaxf(mold, rmax);
                float al   = __expf(mold - mnew);
                float p0   = __expf(a0 - mnew);
                float p1   = __expf(a1 - mnew);
                float rs   = p0 + p1;
#pragma unroll
                for (int msk = 1; msk < 16; msk <<= 1)
                    rs += __shfl_xor(rs, msk, 32);
                if (l16 == 0) {
                    s_m[m]     = mnew;
                    s_l[m]     = s_l[m] * al + rs;
                    s_alpha[m] = al;
                }
                s_pt[m * 32 + l16]      = (__bf16)p0;
                s_pt[m * 32 + 16 + l16] = (__bf16)p1;
            }
        }
        __syncthreads();

        // ---- rescale O by per-row alpha ----
        float av[8];
#pragma unroll
        for (int v = 0; v < 8; ++v) av[v] = s_alpha[qt * 16 + v + hl * 8];
#pragma unroll
        for (int t = 0; t < 16; ++t)
#pragma unroll
            for (int v = 0; v < 8; ++v) o[t][v] *= av[v];

        // ---- read shared P tile as A fragment ----
        v8u pu;
        {
            const uint32_t* pw = (const uint32_t*)(s_pt + qt * 512);
            const int M = l16, kb2 = hl * 4;         // dword offset in row
#pragma unroll
            for (int j = 0; j < 4; ++j) pu[j]     = pw[M * 16 + kb2 + j];
#pragma unroll
            for (int j = 0; j < 4; ++j) pu[4 + j] = pw[M * 16 + 8 + kb2 + j];
        }
        const v16bf pa = as_bf(pu);

        // ---- O += P @ V (pipelined; V transposed: contiguous keys/lane) ----
        {
            const uint16_t* vbase = (const uint16_t*)Vt
                + (size_t)(d0 + l16) * NP + j0 + hl * 8;
            v8u buc = ld_b256_u16(vbase);
#pragma unroll
            for (int t = 0; t < 15; ++t) {
                v8u bun = ld_b256_u16(vbase + (size_t)(t + 1) * 16 * NP);
                o[t] = WMMA_BF16(pa, as_bf(buc), o[t]);
                buc = bun;
            }
            o[15] = WMMA_BF16(pa, as_bf(buc), o[15]);
        }
    }

    // ---- normalize and store ----
    float inv[8];
#pragma unroll
    for (int v = 0; v < 8; ++v) inv[v] = 1.0f / s_l[qt * 16 + v + hl * 8];
#pragma unroll
    for (int t = 0; t < 16; ++t) {
#pragma unroll
        for (int v = 0; v < 8; ++v) {
            int row = q0 + qt * 16 + v + hl * 8;
            if (row < NTOK)
                out[(size_t)row * DIM + d0 + t * 16 + l16] = o[t][v] * inv[v];
        }
    }
}

// ---------------------------------------------------------------------------
extern "C" void kernel_launch(void* const* d_in, const int* in_sizes, int n_in,
                              void* d_out, int out_size, void* d_ws, size_t ws_size,
                              hipStream_t stream) {
    const float* x    = (const float*)d_in[0];
    const float* phi3 = (const float*)d_in[1];
    const float* phis = (const float*)d_in[2];
    const float* phie = (const float*)d_in[3];
    // d_in[4] = delta_pos (unused, as in reference)
    const float* WQ   = (const float*)d_in[5];
    const float* WK   = (const float*)d_in[6];
    const float* WV   = (const float*)d_in[7];
    float* out = (float*)d_out;

    char* ws = (char*)d_ws;
    const size_t seg  = (size_t)NP * DIM * sizeof(uint16_t);   // 16.8 MB
    __bf16* Qb = (__bf16*)(ws);
    __bf16* Kb = (__bf16*)(ws + seg);
    __bf16* Vt = (__bf16*)(ws + 2 * seg);
    __bf16* xb = (__bf16*)(ws + 3 * seg);
    __bf16* Wb = (__bf16*)(ws + 4 * seg);

    cvt_x_kernel<<<(NP * DIM / 8) / 256, 256, 0, stream>>>(x, xb);
    cvt_w_kernel<<<(3 * DIM * DIM / 8) / 256, 256, 0, stream>>>(WQ, WK, WV, Wb);

    dim3 g1(DIM / 256, NP / 16, 3);           // (4, 514, 3)
    qkv_kernel<<<g1, 128, 0, stream>>>(xb, Wb, Qb, Kb, Vt);

    dim3 g2((NTOK + 63) / 64, 1, 1);          // 129 blocks, 64 query rows each
    attn_kernel<<<g2, 512, 0, stream>>>(Qb, Kb, Vt, phi3, phis, phie, out);
}
